// guided_moe_basic_31550829756581
// MI455X (gfx1250) — matrix operand order repaired and checked by reading, not compiled, not run on hardware
//
#include <hip/hip_runtime.h>
#include <hip/hip_bf16.h>
#include <math.h>

// ---------- problem constants ----------
#define BB     8
#define LL     128
#define HH     768
#define EE     7
#define NEXP   4
#define D1     776          // per-position embedding dim (768 + 7 + 1)
#define KP     800          // padded to multiple of 32 for WMMA K-steps
#define NPOS   1024         // B*L
#define PPERB  8256         // L*(L+1)/2
#define NPAIR  66048        // B * PPERB
#define HID    256

// ---------- workspace layout (bytes) ----------
#define OFF_C   0u                               // f16 [1024][800]
#define OFF_WT  1638400u                         // f16 [8][256][800]
#define OFF_GT  4915200u                         // f16 [16][800]
#define OFF_GL  4940800u                         // f32 [1024][16]
#define OFF_H   5006336u                         // f32 [8][1024][256]
#define OFF_O   13394944u                        // f32 [8][1024][2]
#define OFF_AM  13460480u                        // i32 [1024]
#define OFF_OB  13464576u                        // f32 [8]

typedef __attribute__((ext_vector_type(16))) _Float16 v16h;
typedef __attribute__((ext_vector_type(8)))  _Float16 v8h;
typedef __attribute__((ext_vector_type(8)))  float    v8f;

// ---------------------------------------------------------------------------
// Kernel 0: convert + transpose weights to f16, K-padded to 800.
//   WT[g][n][k]: g=0..3 -> exp_W1[e][0:776][n]; g=4..7 -> exp_W1[e][776:][n]
//   GT[n][k]:    n=0..3 -> gate_W[k][n]; n=4..7 -> gate_W[776+k][n-4]; else 0
// ---------------------------------------------------------------------------
__global__ void pack_weights_kernel(const float* __restrict__ expW1,
                                    const float* __restrict__ gateW,
                                    _Float16* __restrict__ WT,
                                    _Float16* __restrict__ GT) {
  int gid = blockIdx.x * 256 + threadIdx.x;
  const int NW = NEXP * HID * KP;              // 819200
  if (gid < NW) {
    int e = gid / (HID * KP);
    int r = gid - e * (HID * KP);
    int n = r / KP;
    int k = r - n * KP;
    float va = 0.f, vb = 0.f;
    if (k < D1) {
      const float* We = expW1 + (long)e * 1552 * HID;
      va = We[(long)k * HID + n];
      vb = We[(long)(D1 + k) * HID + n];
    }
    WT[((long)e * HID + n) * KP + k]          = (_Float16)va;
    WT[((long)(NEXP + e) * HID + n) * KP + k] = (_Float16)vb;
  } else if (gid < NW + 16 * KP) {
    int r = gid - NW;
    int n = r / KP;
    int k = r - n * KP;
    float v = 0.f;
    if (k < D1) {
      if (n < 4)      v = gateW[(long)k * 4 + n];
      else if (n < 8) v = gateW[(long)(D1 + k) * 4 + (n - 4)];
    }
    GT[(long)n * KP + k] = (_Float16)v;
  }
}

// ---------------------------------------------------------------------------
// Kernel 1: per-position work. One block per position p = b*L + l.
//   - emotion logits (exact f32) -> d_out[0:7168] ; argmax -> ws
//   - padded f16 embedding row c[p] = [utter(768) | emo(7) | spk(1) | 0(24)]
// ---------------------------------------------------------------------------
__global__ void prep_pos_kernel(const float* __restrict__ utter,
                                const int*   __restrict__ spk,
                                const float* __restrict__ emoW,
                                const float* __restrict__ emoB,
                                float*       __restrict__ outEmo,
                                _Float16*    __restrict__ Cpad,
                                int*         __restrict__ argm) {
  __shared__ float u[HH];
  __shared__ float ep[EE];
  const int p = blockIdx.x;
  const int tid = threadIdx.x;
  const float* up = utter + (long)p * HH;
  for (int i = tid; i < HH; i += 256) u[i] = up[i];
  __syncthreads();
  const int wid = tid >> 5, lane = tid & 31;
  if (wid < EE) {
    float s = 0.f;
    for (int i = lane; i < HH; i += 32) s += u[i] * emoW[i * EE + wid];
#pragma unroll
    for (int off = 16; off > 0; off >>= 1) s += __shfl_xor(s, off, 32);
    if (lane == 0) ep[wid] = s + emoB[wid];
  }
  __syncthreads();
  _Float16* crow = Cpad + (long)p * KP;
  for (int i = tid; i < HH; i += 256) crow[i] = (_Float16)u[i];
  if (tid < EE) {
    float v = ep[tid];
    outEmo[p * EE + tid] = v;
    crow[HH + tid] = (_Float16)v;
  }
  if (tid == 7)  crow[D1 - 1] = (_Float16)(float)spk[p];
  if (tid >= 8 && tid < 32) crow[D1 + (tid - 8)] = (_Float16)0.f;  // pad 776..799
  if (tid == 32) {
    int best = 0; float bv = ep[0];
    for (int j = 1; j < EE; ++j) if (ep[j] > bv) { bv = ep[j]; best = j; }
    argm[p] = best;
  }
}

// ---------------------------------------------------------------------------
// Kernel 2: register-tiled WMMA GEMM
//   Out[g] = A(f16,[1024][800]) x B(g)(f16,[800][N]), B pre-transposed (BT[n][k]).
//   Each wave owns an (MT*16) x (NT*16) output block: A fragments are reused
//   across NT column-tiles and B fragments across MT row-tiles, so per K-step
//   we issue 2*MT + 2*NT b128 loads for MT*NT v_wmma_f32_16x16x32_f16 ops
//   (1.5 loads/WMMA at MT=2,NT=4 vs 4 for the untiled version).
//   blockDim=128 (4 full waves; EXEC all ones, required by WMMA).
//   Fragment layout per CDNA5 ISA wave32 tables:
//     A: lane (h=l>>4,m=l&15): halfs [k0+8h .. +7] and [k0+16+8h .. +7]
//     B: lane (h,n): halfs [k0+8h .. +7] and [k0+16+8h .. +7] of column n
//     D: lane (h,n): VGPR r holds row 8h+r, col n
// ---------------------------------------------------------------------------
template <int MT, int NT>
__global__ void wmma_gemm_kernel(const _Float16* __restrict__ A,
                                 const _Float16* __restrict__ BT,
                                 float*          __restrict__ Out,
                                 int ldOut, long strideB, long strideOut) {
  const int wid  = threadIdx.x >> 5;
  const int lane = threadIdx.x & 31;
  const int hf   = lane >> 4;
  const int sub  = lane & 15;
  const int tileM0 = (blockIdx.x * 4 + wid) * MT;   // row tile index (16-row units)
  const int tileN0 = blockIdx.y * NT;               // col tile index (16-col units)

  const _Float16* a0[MT];
  const _Float16* b0[NT];
#pragma unroll
  for (int i = 0; i < MT; ++i)
    a0[i] = A + (long)((tileM0 + i) * 16 + sub) * KP;
#pragma unroll
  for (int j = 0; j < NT; ++j)
    b0[j] = BT + (long)blockIdx.z * strideB + (long)((tileN0 + j) * 16 + sub) * KP;
  float* out = Out + (long)blockIdx.z * strideOut;

  v8f c[MT][NT];
#pragma unroll
  for (int i = 0; i < MT; ++i)
#pragma unroll
    for (int j = 0; j < NT; ++j) c[i][j] = (v8f){};

  for (int k0 = 0; k0 < KP; k0 += 32) {
    union F { v16h v; v8h h[2]; };
    F a[MT], b[NT];
#pragma unroll
    for (int i = 0; i < MT; ++i) {
      a[i].h[0] = *(const v8h*)(a0[i] + k0 + 8 * hf);
      a[i].h[1] = *(const v8h*)(a0[i] + k0 + 16 + 8 * hf);
    }
#pragma unroll
    for (int j = 0; j < NT; ++j) {
      b[j].h[0] = *(const v8h*)(b0[j] + k0 + 8 * hf);
      b[j].h[1] = *(const v8h*)(b0[j] + k0 + 16 + 8 * hf);
    }
#pragma unroll
    for (int i = 0; i < MT; ++i)
#pragma unroll
      for (int j = 0; j < NT; ++j)
        c[i][j] = __builtin_amdgcn_wmma_f32_16x16x32_f16(
            false, a[i].v, false, b[j].v, (short)0, c[i][j], false, false);
  }

#pragma unroll
  for (int i = 0; i < MT; ++i) {
    const int rowBase = (tileM0 + i) * 16 + 8 * hf;
#pragma unroll
    for (int j = 0; j < NT; ++j) {
      const int col = (tileN0 + j) * 16 + sub;
#pragma unroll
      for (int r = 0; r < 8; ++r)
        out[(long)(rowBase + r) * ldOut + col] = c[i][j][r];
    }
  }
}

// ---------------------------------------------------------------------------
// Kernel 3: tiny second-stage contraction O[g][p][c] = H[g][p][:] @ W2[e][:,c]
//   plus ob[e][c] = b1[e] @ W2[e][:,c] + b2[e][c]
// ---------------------------------------------------------------------------
__global__ void stage2_kernel(const float* __restrict__ Hbuf,
                              const float* __restrict__ W2,
                              const float* __restrict__ b1,
                              const float* __restrict__ b2,
                              float* __restrict__ Obuf,
                              float* __restrict__ ob) {
  int gid = blockIdx.x * 256 + threadIdx.x;
  if (gid < 8 * NPOS) {
    int g = gid >> 10;
    int p = gid & (NPOS - 1);
    int e = g & 3;
    const float* h = Hbuf + ((long)g * NPOS + p) * HID;
    const float* w = W2 + (long)e * HID * 2;
    float a0 = 0.f, a1 = 0.f;
    for (int n = 0; n < HID; ++n) {
      float hv = h[n];
      a0 += hv * w[n * 2 + 0];
      a1 += hv * w[n * 2 + 1];
    }
    Obuf[((long)g * NPOS + p) * 2 + 0] = a0;
    Obuf[((long)g * NPOS + p) * 2 + 1] = a1;
  }
  if (gid < 8) {
    int e = gid >> 1, cc = gid & 1;
    const float* w  = W2 + (long)e * HID * 2;
    const float* bb = b1 + (long)e * HID;
    float s = b2[e * 2 + cc];
    for (int n = 0; n < HID; ++n) s += bb[n] * w[n * 2 + cc];
    ob[gid] = s;
  }
}

// ---------------------------------------------------------------------------
// Kernel 4: per-pair finalize (softmax gate + guided blend + expert mix)
//   tril_indices order: pair i -> end = tri-row(i), t = i - end(end+1)/2
// ---------------------------------------------------------------------------
__global__ void finalize_kernel(const int*   __restrict__ spk,
                                const int*   __restrict__ argm,
                                const float* __restrict__ GL,
                                const float* __restrict__ gateB,
                                const float* __restrict__ Obuf,
                                const float* __restrict__ ob,
                                float*       __restrict__ outCause) {
  int gid = blockIdx.x * 256 + threadIdx.x;
  if (gid >= NPAIR) return;
  int b = gid / PPERB;
  int i = gid - b * PPERB;
  int end = (int)((sqrtf(8.0f * (float)i + 1.0f) - 1.0f) * 0.5f);
  while ((end + 1) * (end + 2) / 2 <= i) ++end;
  while (end * (end + 1) / 2 > i)       --end;
  int t  = i - end * (end + 1) / 2;
  int pt = b * LL + t;
  int pe = b * LL + end;

  bool s_eq = spk[pt] == spk[pe];
  bool e_eq = argm[pt] == argm[pe];
  int lbl = (s_eq && e_eq) ? 0 : (s_eq ? 1 : (e_eq ? 2 : 3));

  float lg[NEXP];
  float mx = -1e30f;
#pragma unroll
  for (int e = 0; e < NEXP; ++e) {
    lg[e] = GL[pt * 16 + e] + GL[pe * 16 + 4 + e] + gateB[e];
    mx = fmaxf(mx, lg[e]);
  }
  float se = 0.f;
#pragma unroll
  for (int e = 0; e < NEXP; ++e) { lg[e] = __expf(lg[e] - mx); se += lg[e]; }
  float inv = 1.0f / se;

  float o0 = 0.f, o1 = 0.f;
#pragma unroll
  for (int e = 0; e < NEXP; ++e) {
    float gate = 0.6f * ((e == lbl) ? 1.0f : 0.0f) + 0.4f * lg[e] * inv;
    o0 += gate * (Obuf[((long)e * NPOS + pt) * 2 + 0]
                + Obuf[((long)(NEXP + e) * NPOS + pe) * 2 + 0] + ob[e * 2 + 0]);
    o1 += gate * (Obuf[((long)e * NPOS + pt) * 2 + 1]
                + Obuf[((long)(NEXP + e) * NPOS + pe) * 2 + 1] + ob[e * 2 + 1]);
  }
  outCause[(long)gid * 2 + 0] = o0;
  outCause[(long)gid * 2 + 1] = o1;
}

// ---------------------------------------------------------------------------
extern "C" void kernel_launch(void* const* d_in, const int* in_sizes, int n_in,
                              void* d_out, int out_size, void* d_ws, size_t ws_size,
                              hipStream_t stream) {
  const float* utter = (const float*)d_in[0];
  const int*   spk   = (const int*)  d_in[1];
  const float* emoW  = (const float*)d_in[2];
  const float* emoB  = (const float*)d_in[3];
  const float* gateW = (const float*)d_in[4];
  const float* gateB = (const float*)d_in[5];
  const float* expW1 = (const float*)d_in[6];
  const float* expB1 = (const float*)d_in[7];
  const float* expW2 = (const float*)d_in[8];
  const float* expB2 = (const float*)d_in[9];
  float* out = (float*)d_out;

  char* ws = (char*)d_ws;
  _Float16* Cpad = (_Float16*)(ws + OFF_C);
  _Float16* WT   = (_Float16*)(ws + OFF_WT);
  _Float16* GT   = (_Float16*)(ws + OFF_GT);
  float*    GL   = (float*)   (ws + OFF_GL);
  float*    Hbuf = (float*)   (ws + OFF_H);
  float*    Obuf = (float*)   (ws + OFF_O);
  int*      argm = (int*)     (ws + OFF_AM);
  float*    obv  = (float*)   (ws + OFF_OB);

  // 0) pack weights (f16, transposed, K-padded)
  {
    int total = NEXP * HID * KP + 16 * KP;      // 832000
    pack_weights_kernel<<<(total + 255) / 256, 256, 0, stream>>>(expW1, gateW, WT, GT);
  }
  // 1) per-position embedding + emotion head
  prep_pos_kernel<<<NPOS, 256, 0, stream>>>(utter, spk, emoW, emoB, out, Cpad, argm);
  // 2a) 8 expert-half GEMMs [1024x800]x[800x256] via register-tiled WMMA
  //     wave block = 32x64 output (MT=2, NT=4); grid: Mgroups=8, Ngroups=4, g=8
  wmma_gemm_kernel<2, 4><<<dim3(8, 4, 8), 128, 0, stream>>>(
      Cpad, WT, Hbuf, HID, (long)HID * KP, (long)NPOS * HID);
  // 2b) gate GEMM [1024x800]x[800x16] (single 16-wide tile per wave)
  wmma_gemm_kernel<1, 1><<<dim3(16, 1, 1), 128, 0, stream>>>(
      Cpad, GT, GL, 16, 0L, 0L);
  // 3) H @ W2 (+ bias term)
  stage2_kernel<<<(8 * NPOS + 255) / 256, 256, 0, stream>>>(
      Hbuf, expW2, expB1, expB2, Obuf, obv);
  // 4) per-pair gate/blend/mix -> cause_pred
  finalize_kernel<<<(NPAIR + 255) / 256, 256, 0, stream>>>(
      spk, argm, GL, gateB, Obuf, obv, out + NPOS * EE);
}
